// HierarchyGCNModule_32238024524217
// MI455X (gfx1250) — matrix operand order repaired and checked by reading, not compiled, not run on hardware
//
#include <hip/hip_runtime.h>
#include <hip/hip_bf16.h>
#include <math.h>

// ---------------------------------------------------------------------------
// HierarchyGCN forward for MI455X (gfx1250, wave32, WMMA).
//   B=4096, N=103, D=300.
//   out[b,n,d] = relu( g_in[b,n] *(adj_in @x_b + eb)[n,d]
//                    + g_out[b,n]*(adj_out@x_b + oeb)[n,d]
//                    + g_loop[b,n]* x[b,n,d] )
// One workgroup per batch element. GEMMs in f16 via v_wmma_f32_16x16x32_f16
// (f32 accumulate). Register-blocked 2 D-tiles x 2 adj matrices per wave
// iteration: 8 ds_load_b128 feed 4 WMMAs (2 loads/WMMA).
// Roofline: ~1 GB HBM traffic (x read + out write) ~= 43us @ 23.3 TB/s;
// 52 GFLOP of f16 WMMA hides underneath.
// ---------------------------------------------------------------------------

typedef _Float16 half8 __attribute__((ext_vector_type(8)));
typedef _Float16 v16h  __attribute__((ext_vector_type(16)));
typedef float    v8f   __attribute__((ext_vector_type(8)));

#define NB    103              // graph nodes (GEMM M and K)
#define DD    300              // feature dim  (GEMM N)
#define KPAD  136              // padded K stride in halves (272B rows, 16B aligned)
#define MROWS 112              // M padded to 7*16
#define DPAD  320              // D padded to 20*16 (even # of tiles for pairing)
#define MT    7                // M tiles
#define DT2   10               // D tile *pairs*
#define NTHREADS 256

// A fragment (16x32 f16, MxK). ISA 7.12.2: lane L holds row M=L%16.
// halves 0..7  = K {k0 .. k0+7},  halves 8..15 = K {k0+16 .. k0+23},
// with k0 = kk + 8*(lane>=16). Two 16B contiguous LDS chunks.
__device__ __forceinline__ v16h load_frag_a(const _Float16* base, int mt, int kk, int lane) {
    int m  = mt * 16 + (lane & 15);
    int k0 = kk + ((lane >> 4) << 3);
    const _Float16* p = base + m * KPAD + k0;
    union { half8 h2[2]; v16h v; } u;
    u.h2[0] = *(const half8*)(p);
    u.h2[1] = *(const half8*)(p + 16);
    return u.v;
}

// B fragment (32x16 f16, KxN). Lane L holds column N=L%16;
// halves 0..7 = K {kb .. kb+7}, halves 8..15 = K {kb+16 .. kb+23}, kb = kk + 8*(lane>=16).
// xs is stored d-major [DPAD][KPAD] so this is two contiguous 16B row reads.
__device__ __forceinline__ v16h load_frag_b(const _Float16* xs, int kk, int dt, int lane) {
    int d  = dt * 16 + (lane & 15);
    int kb = kk + ((lane >> 4) << 3);
    const _Float16* p = xs + d * KPAD + kb;
    union { half8 h2[2]; v16h v; } u;
    u.h2[0] = *(const half8*)(p);
    u.h2[1] = *(const half8*)(p + 16);
    return u.v;
}

__global__ __launch_bounds__(NTHREADS)
void hgcn_fwd_kernel(const float* __restrict__ x,
                     const float* __restrict__ adj_in,
                     const float* __restrict__ adj_out,
                     const float* __restrict__ edge_bias,
                     const float* __restrict__ gate_w,
                     const float* __restrict__ bias_gate,
                     const float* __restrict__ out_edge_bias,
                     const float* __restrict__ out_gate_w,
                     const float* __restrict__ out_bias_gate,
                     const float* __restrict__ loop_gate,
                     float* __restrict__ out) {
    // LDS staging (~150 KB -> 2 workgroups per 320KB WGP)
    __shared__ __attribute__((aligned(16))) _Float16 xs_t[DPAD * KPAD];   // x_b^T as f16: [d][k]
    __shared__ __attribute__((aligned(16))) _Float16 aI [MROWS * KPAD];   // adj_in  f16, zero-padded
    __shared__ __attribute__((aligned(16))) _Float16 aO [MROWS * KPAD];   // adj_out f16, zero-padded
    __shared__ float wbuf[3 * DPAD];                                      // 3 gate weight vectors
    __shared__ float gl[3 * MROWS];                                       // per-row gates

    const int tid  = threadIdx.x;
    const int b    = blockIdx.x;
    const float* xb = x + (size_t)b * NB * DD;

    // ---- Phase 1: stage adj matrices (f16, zero K/M padding) ----
    for (int i = tid; i < MROWS * KPAD; i += NTHREADS) {
        int m = i / KPAD, k = i - m * KPAD;
        bool ok = (m < NB) & (k < NB);
        aI[i] = ok ? (_Float16)adj_in [m * NB + k] : (_Float16)0.f;
        aO[i] = ok ? (_Float16)adj_out[m * NB + k] : (_Float16)0.f;
    }
    // gate weight vectors -> LDS
    for (int i = tid; i < DPAD; i += NTHREADS) {
        bool ok = i < DD;
        wbuf[i]            = ok ? gate_w[i]      : 0.f;
        wbuf[DPAD + i]     = ok ? out_gate_w[i]  : 0.f;
        wbuf[2 * DPAD + i] = ok ? loop_gate[i]   : 0.f;
    }
    // x_b -> f16 transposed into LDS (coalesced global reads over d)
    for (int i = tid; i < NB * DPAD; i += NTHREADS) {
        int k = i / DPAD, d = i - k * DPAD;
        float v = (d < DD) ? xb[k * DD + d] : 0.f;
        xs_t[d * KPAD + k] = (_Float16)v;
    }
    // zero the contraction pad rows k = 103..135 (required for WMMA K-padding)
    for (int i = tid; i < DPAD * (KPAD - NB); i += NTHREADS) {
        int d = i / (KPAD - NB), k = NB + (i - d * (KPAD - NB));
        xs_t[d * KPAD + k] = (_Float16)0.f;
    }
    __syncthreads();

    // ---- Phase 2: per-row sigmoid gates (103 rows, dot over D from LDS) ----
    if (tid < NB) {
        float s0 = 0.f, s1 = 0.f, s2 = 0.f;
        for (int d = 0; d < DD; ++d) {
            float xv = (float)xs_t[d * KPAD + tid];
            s0 = fmaf(xv, wbuf[d],            s0);
            s1 = fmaf(xv, wbuf[DPAD + d],     s1);
            s2 = fmaf(xv, wbuf[2 * DPAD + d], s2);
        }
        s0 += bias_gate[tid];
        s1 += out_bias_gate[tid];
        gl[tid]             = 1.f / (1.f + __expf(-s0));
        gl[MROWS + tid]     = 1.f / (1.f + __expf(-s1));
        gl[2 * MROWS + tid] = 1.f / (1.f + __expf(-s2));
    }
    __syncthreads();

    // ---- Phase 3: WMMA GEMMs (2 adj x 2 d-tiles per unit) + epilogue ----
    const int wv   = tid >> 5;
    const int lane = tid & 31;
    float* outb = out + (size_t)b * NB * DD;

    for (int u = wv; u < MT * DT2; u += NTHREADS / 32) {
        const int mt  = u / DT2;
        const int dp  = u - mt * DT2;
        const int dt0 = 2 * dp;
        const int dt1 = 2 * dp + 1;

        v8f c00 = {}, c01 = {}, c10 = {}, c11 = {};
        #pragma unroll
        for (int ks = 0; ks < 4; ++ks) {
            const int kk = ks * 32;
            v16h b0 = load_frag_b(xs_t, kk, dt0, lane);
            v16h b1 = load_frag_b(xs_t, kk, dt1, lane);
            v16h ai = load_frag_a(aI, mt, kk, lane);
            v16h ao = load_frag_a(aO, mt, kk, lane);
            c00 = __builtin_amdgcn_wmma_f32_16x16x32_f16(
                false, ai, false, b0, (short)0, c00, false, false);
            c01 = __builtin_amdgcn_wmma_f32_16x16x32_f16(
                false, ai, false, b1, (short)0, c01, false, false);
            c10 = __builtin_amdgcn_wmma_f32_16x16x32_f16(
                false, ao, false, b0, (short)0, c10, false, false);
            c11 = __builtin_amdgcn_wmma_f32_16x16x32_f16(
                false, ao, false, b1, (short)0, c11, false, false);
        }

        // C/D layout: VGPR v -> M = v + 8*(lane>=16); N = lane&15
        const int mbase = mt * 16 + ((lane >> 4) << 3);
        const int d0    = dt0 * 16 + (lane & 15);
        const int d1    = dt1 * 16 + (lane & 15);

        #pragma unroll
        for (int v = 0; v < 8; ++v) {
            const int m = mbase + v;
            if (m < NB) {
                const float gi = gl[m];
                const float go = gl[MROWS + m];
                const float gp = gl[2 * MROWS + m];
                if (d0 < DD) {
                    float xv  = (float)xs_t[d0 * KPAD + m];
                    int   off = m * DD + d0;
                    float val = gi * (c00[v] + edge_bias[off])
                              + go * (c10[v] + out_edge_bias[off])
                              + gp * xv;
                    outb[off] = val > 0.f ? val : 0.f;
                }
                if (d1 < DD) {
                    float xv  = (float)xs_t[d1 * KPAD + m];
                    int   off = m * DD + d1;
                    float val = gi * (c01[v] + edge_bias[off])
                              + go * (c11[v] + out_edge_bias[off])
                              + gp * xv;
                    outb[off] = val > 0.f ? val : 0.f;
                }
            }
        }
    }
}

extern "C" void kernel_launch(void* const* d_in, const int* in_sizes, int n_in,
                              void* d_out, int out_size, void* d_ws, size_t ws_size,
                              hipStream_t stream) {
    const float* x             = (const float*)d_in[0];
    const float* adj_in        = (const float*)d_in[1];
    const float* adj_out       = (const float*)d_in[2];
    const float* edge_bias     = (const float*)d_in[3];
    const float* gate_w        = (const float*)d_in[4];
    const float* bias_gate     = (const float*)d_in[5];
    const float* out_edge_bias = (const float*)d_in[6];
    const float* out_gate_w    = (const float*)d_in[7];
    const float* out_bias_gate = (const float*)d_in[8];
    const float* loop_gate     = (const float*)d_in[9];
    float* out = (float*)d_out;

    const int B = 4096;
    hgcn_fwd_kernel<<<B, NTHREADS, 0, stream>>>(
        x, adj_in, adj_out, edge_bias, gate_w, bias_gate,
        out_edge_bias, out_gate_w, out_bias_gate, loop_gate, out);
}
